// FastSpectralAttention_4621384811001
// MI455X (gfx1250) — compile-verified, matrix operand order checked by reading, not run on hardware
//
#include <hip/hip_runtime.h>

// ---------------- problem constants ----------------
#define B_    2
#define T_    4096
#define C_    1024
#define H_    8
#define K_    32
#define HK_   256      // H_*K_
#define NQKV_ 768      // 3*H_*K_
#define MROWS_ 8192    // B_*T_
#define L_    64       // scan chunk length
#define NC_   64       // T_/L_
#define BH_   16       // B_*H_

typedef __attribute__((ext_vector_type(16))) __bf16 v16bf;
typedef __attribute__((ext_vector_type(8)))  __bf16 v8bf;
typedef __attribute__((ext_vector_type(8)))  float  v8f;
typedef __attribute__((ext_vector_type(4)))  int    v4i;

#define AS1 __attribute__((address_space(1)))
#define AS3 __attribute__((address_space(3)))

#if __has_builtin(__builtin_amdgcn_global_load_async_to_lds_b128)
#define HAVE_ASYNC 1
// gp: generic pointer into global memory; lp: generic pointer into LDS
#define GASYNC_B128(gp, lp)                                                   \
  __builtin_amdgcn_global_load_async_to_lds_b128(                             \
      (AS1 v4i*)(AS1 void*)(gp), (AS3 v4i*)(AS3 void*)(lp), 0, 0)
#else
#define HAVE_ASYNC 0
#endif

#if __has_builtin(__builtin_amdgcn_s_wait_asynccnt)
#define WAIT_ASYNC(n) __builtin_amdgcn_s_wait_asynccnt(n)
#else
#define WAIT_ASYNC(n) asm volatile("s_wait_asynccnt %0" ::"i"(n) : "memory")
#endif

static __device__ __forceinline__ v8f wmma_bf16(v16bf a, v16bf b, v8f c) {
  // D = A(16x32,bf16) * B(32x16,bf16) + C(16x16,f32)
  return __builtin_amdgcn_wmma_f32_16x16x32_bf16(false, a, false, b, (short)0, c,
                                                 false, false);
}

// k index for VGPR-pair j (0..7) given lane-half g (ISA 7.12.2 16-bit layout)
static __device__ __forceinline__ int frag_k0(int j, int g) {
  return ((j & 4) << 2) + 8 * g + 2 * (j & 3);
}

// Fragment whose element [2j+e] = row[frag_k0(j,g)+e]; row must be 16B aligned.
// Covers: A row-major (row = A row m), B from a transposed/row-of-K source
// (row = source row n). Two 16-byte loads, no conversions.
static __device__ __forceinline__ v16bf frag_row32(const __bf16* row, int g) {
  v8bf lo = *(const v8bf*)(row + 8 * g);        // k = 8g .. 8g+7
  v8bf hi = *(const v8bf*)(row + 16 + 8 * g);   // k = 16+8g .. 16+8g+7
  v16bf a;
#pragma unroll
  for (int j = 0; j < 8; ++j) { a[j] = lo[j]; a[j + 8] = hi[j]; }
  return a;
}

// A fragment (16x32) from transposed bf16 source: A[m][k] = src[k*ld + m]
static __device__ __forceinline__ v16bf load_A_cm_bf16(const __bf16* src, int ld) {
  int lane = threadIdx.x & 31;
  int m = lane & 15, g = lane >> 4;
  v16bf a;
#pragma unroll
  for (int j = 0; j < 8; ++j) {
    int k0 = frag_k0(j, g);
    a[2 * j]     = src[k0 * ld + m];
    a[2 * j + 1] = src[(k0 + 1) * ld + m];
  }
  return a;
}

// B fragment (32x16) from row-major bf16 source: B[k][n] = src[k*ld + n]
static __device__ __forceinline__ v16bf load_B_rm_bf16(const __bf16* src, int ld) {
  int lane = threadIdx.x & 31;
  int n = lane & 15, g = lane >> 4;
  v16bf b;
#pragma unroll
  for (int j = 0; j < 8; ++j) {
    int k0 = frag_k0(j, g);
    b[2 * j]     = src[k0 * ld + n];
    b[2 * j + 1] = src[(k0 + 1) * ld + n];
  }
  return b;
}

// B fragment (32x16) from row-major f32 source (converted)
static __device__ __forceinline__ v16bf load_B_rm_f32(const float* src, int ld) {
  int lane = threadIdx.x & 31;
  int n = lane & 15, g = lane >> 4;
  v16bf b;
#pragma unroll
  for (int j = 0; j < 8; ++j) {
    int k0 = frag_k0(j, g);
    b[2 * j]     = (__bf16)src[k0 * ld + n];
    b[2 * j + 1] = (__bf16)src[(k0 + 1) * ld + n];
  }
  return b;
}

static __device__ __forceinline__ void store_D_f32(float* dst, int ld, v8f c) {
  int lane = threadIdx.x & 31;
  int n = lane & 15, g = lane >> 4;
#pragma unroll
  for (int j = 0; j < 8; ++j) dst[(j + 8 * g) * ld + n] = c[j];
}

static __device__ __forceinline__ void store_D_bf16(__bf16* dst, int ld, v8f c) {
  int lane = threadIdx.x & 31;
  int n = lane & 15, g = lane >> 4;
#pragma unroll
  for (int j = 0; j < 8; ++j) dst[(j + 8 * g) * ld + n] = (__bf16)c[j];
}

// ---------------------------------------------------------------------------
// Kernel 0: elementwise f32 -> bf16 (vectorized by 8)
// ---------------------------------------------------------------------------
__global__ __launch_bounds__(256) void k_cvt(const float* __restrict__ src,
                                             __bf16* __restrict__ dst, int n8) {
  int i = blockIdx.x * 256 + threadIdx.x;
  if (i >= n8) return;
  v8f v = *(const v8f*)(src + (size_t)i * 8);
  v8bf o;
#pragma unroll
  for (int j = 0; j < 8; ++j) o[j] = (__bf16)v[j];
  *(v8bf*)(dst + (size_t)i * 8) = o;
}

// ---------------------------------------------------------------------------
// Kernel 1: qkv = x @ W_qkv^T (bf16 tiles, async-staged LDS, 64x128 block),
// fused epilogue: scatter into q/k/v (B,H,T,K) bf16, q *= sb*K^-0.5, k *= sb.
// ---------------------------------------------------------------------------
__global__ __launch_bounds__(256) void k_qkv(const __bf16* __restrict__ xb,
                                             const float* __restrict__ sb,
                                             const __bf16* __restrict__ Wq,
                                             __bf16* __restrict__ qb,
                                             __bf16* __restrict__ kb,
                                             __bf16* __restrict__ vb) {
  __shared__ __align__(16) __bf16 Abf[2][64][40];
  __shared__ __align__(16) __bf16 Bbf[2][128][40];
  const int m0 = blockIdx.x * 64;
  const int n0 = blockIdx.y * 128;
  const int tid = threadIdx.x;
  const int wave = tid >> 5, lane = tid & 31;
  const int m = lane & 15, g = lane >> 4;
  const int mt = wave & 3;
  const int ntBase = (wave >> 2) * 4;
  const int ar = tid >> 2, ac = (tid & 3) * 8;   // A: 256 chunks of 8 bf16
  const v8f vz = {0.f, 0.f, 0.f, 0.f, 0.f, 0.f, 0.f, 0.f};
  v8f acc[4] = {vz, vz, vz, vz};
  const int NK = C_ / 32;

#if HAVE_ASYNC
  auto issue_tile = [&](int ks) {
    int kk = ks * 32, bufi = ks & 1;
    GASYNC_B128(xb + (size_t)(m0 + ar) * C_ + kk + ac, &Abf[bufi][ar][ac]);
#pragma unroll
    for (int i = 0; i < 2; ++i) {
      int cidx = tid + 256 * i;
      int r = cidx >> 2, c8 = (cidx & 3) * 8;
      GASYNC_B128(Wq + (size_t)(n0 + r) * C_ + kk + c8, &Bbf[bufi][r][c8]);
    }
  };
#else
  auto copy_tile = [&](int ks) {
    int kk = ks * 32, bufi = ks & 1;
    *(v8bf*)&Abf[bufi][ar][ac] =
        *(const v8bf*)(xb + (size_t)(m0 + ar) * C_ + kk + ac);
#pragma unroll
    for (int i = 0; i < 2; ++i) {
      int cidx = tid + 256 * i;
      int r = cidx >> 2, c8 = (cidx & 3) * 8;
      *(v8bf*)&Bbf[bufi][r][c8] =
          *(const v8bf*)(Wq + (size_t)(n0 + r) * C_ + kk + c8);
    }
  };
#endif
  auto compute_tile = [&](int bufi) {
    v16bf a = frag_row32(&Abf[bufi][mt * 16 + m][0], g);
#pragma unroll
    for (int t4 = 0; t4 < 4; ++t4) {
      v16bf b = frag_row32(&Bbf[bufi][(ntBase + t4) * 16 + m][0], g);
      acc[t4] = wmma_bf16(a, b, acc[t4]);
    }
  };

#if HAVE_ASYNC
  issue_tile(0);
  for (int ks = 0; ks < NK; ++ks) {
    if (ks + 1 < NK) { issue_tile(ks + 1); WAIT_ASYNC(3); } else { WAIT_ASYNC(0); }
    __syncthreads();
    compute_tile(ks & 1);
    __syncthreads();
  }
#else
  for (int ks = 0; ks < NK; ++ks) {
    copy_tile(ks);
    __syncthreads();
    compute_tile(ks & 1);
    __syncthreads();
  }
#endif

  const float scale = 0.17677669529663687f;  // 32^-0.5
#pragma unroll
  for (int t4 = 0; t4 < 4; ++t4) {
    int nt = ntBase + t4;
#pragma unroll
    for (int j = 0; j < 8; ++j) {
      int gm = m0 + mt * 16 + j + 8 * g;
      int gn = n0 + nt * 16 + m;
      int b  = gm >> 12;           // / T_
      int t  = gm & (T_ - 1);
      int which = gn >> 8;         // 0=q 1=k 2=v
      int h  = (gn >> 5) & 7;
      int kq = gn & 31;
      float val = acc[t4][j];
      if (which < 2) {
        float s = sb[((size_t)b * T_ + t) * K_ + kq];
        val *= (which == 0) ? s * scale : s;
      }
      __bf16* dst = (which == 0) ? qb : (which == 1) ? kb : vb;
      dst[(((size_t)b * H_ + h) * T_ + t) * K_ + kq] = (__bf16)val;
    }
  }
}

// ---------------------------------------------------------------------------
// Kernel 2: per-chunk state S_c = K_chunk^T @ V_chunk (32x32 f32), chunk/wave
// ---------------------------------------------------------------------------
__global__ __launch_bounds__(256) void k_state(const __bf16* __restrict__ kb,
                                               const __bf16* __restrict__ vb,
                                               float* __restrict__ Sb) {
  int gwave = blockIdx.x * (blockDim.x >> 5) + (threadIdx.x >> 5);  // 0..1023
  int bh = gwave >> 6;
  int c  = gwave & (NC_ - 1);
  const __bf16* kptr = kb + (size_t)bh * T_ * K_ + (size_t)c * L_ * K_;
  const __bf16* vptr = vb + (size_t)bh * T_ * K_ + (size_t)c * L_ * K_;
  float* sptr = Sb + ((size_t)bh * NC_ + c) * (K_ * K_);
  const v8f vz = {0.f, 0.f, 0.f, 0.f, 0.f, 0.f, 0.f, 0.f};
#pragma unroll
  for (int mtv = 0; mtv < 2; ++mtv) {
#pragma unroll
    for (int nt = 0; nt < 2; ++nt) {
      v8f acc = vz;
#pragma unroll
      for (int kt = 0; kt < 2; ++kt) {
        v16bf a = load_A_cm_bf16(kptr + (size_t)kt * 32 * K_ + mtv * 16, K_);
        v16bf b = load_B_rm_bf16(vptr + (size_t)kt * 32 * K_ + nt * 16, K_);
        acc = wmma_bf16(a, b, acc);
      }
      store_D_f32(sptr + (mtv * 16) * K_ + nt * 16, K_, acc);
    }
  }
}

// ---------------------------------------------------------------------------
// Kernel 3: in-place exclusive prefix over the 64 chunk states per (b,h)
// ---------------------------------------------------------------------------
__global__ __launch_bounds__(256) void k_prefix(float* __restrict__ Sb) {
  float* base = Sb + (size_t)blockIdx.x * NC_ * (K_ * K_);
  int tid = threadIdx.x;
  float acc[4] = {0.f, 0.f, 0.f, 0.f};
  for (int c = 0; c < NC_; ++c) {
    float* p = base + c * (K_ * K_);
#pragma unroll
    for (int i = 0; i < 4; ++i) {
      float v = p[tid + 256 * i];
      p[tid + 256 * i] = acc[i];
      acc[i] += v;
    }
  }
}

// ---------------------------------------------------------------------------
// Kernel 4: per chunk: out = causal(Q K^T) V + Q @ P_c -> attn (B,T,H*K) bf16
// ---------------------------------------------------------------------------
__global__ __launch_bounds__(128) void k_attn(const __bf16* __restrict__ qb,
                                              const __bf16* __restrict__ kb,
                                              const __bf16* __restrict__ vb,
                                              const float* __restrict__ Sb,
                                              __bf16* __restrict__ attn) {
  __shared__ __align__(16) __bf16 sc[4][64][72];
  int wv = threadIdx.x >> 5, lane = threadIdx.x & 31;
  int gwave = blockIdx.x * 4 + wv;        // 0..1023
  int bh = gwave >> 6;
  int c  = gwave & (NC_ - 1);
  int b  = bh >> 3, h = bh & 7;
  int t0 = c * L_;
  const __bf16* qptr = qb + (size_t)bh * T_ * K_ + (size_t)t0 * K_;
  const __bf16* kptr = kb + (size_t)bh * T_ * K_ + (size_t)t0 * K_;
  const __bf16* vptr = vb + (size_t)bh * T_ * K_ + (size_t)t0 * K_;
  const float* pptr = Sb + ((size_t)bh * NC_ + c) * (K_ * K_);
  __bf16 (*S)[72] = sc[wv];
  const int n = lane & 15, g = lane >> 4;
  const v8f vz = {0.f, 0.f, 0.f, 0.f, 0.f, 0.f, 0.f, 0.f};

  v16bf aq[4];
#pragma unroll
  for (int ti = 0; ti < 4; ++ti)
    aq[ti] = frag_row32(qptr + (size_t)(ti * 16 + n) * K_, g);

  // scores = causal(Q K^T), staged to LDS (bf16) for re-fragmentation
#pragma unroll
  for (int ti = 0; ti < 4; ++ti) {
#pragma unroll
    for (int tj = 0; tj < 4; ++tj) {
      if (tj > ti) {
#pragma unroll
        for (int j = 0; j < 8; ++j)
          S[ti * 16 + j + 8 * g][tj * 16 + n] = (__bf16)0.f;
      } else {
        // B[k][nn] = K[tj*16+nn][k]  -> row-of-K fragment
        v16bf bk = frag_row32(kptr + (size_t)(tj * 16 + n) * K_, g);
        v8f s = wmma_bf16(aq[ti], bk, vz);
        if (ti == tj) {
#pragma unroll
          for (int j = 0; j < 8; ++j)
            if (n > j + 8 * g) s[j] = 0.f;   // strict upper triangle -> 0
        }
#pragma unroll
        for (int j = 0; j < 8; ++j)
          S[ti * 16 + j + 8 * g][tj * 16 + n] = (__bf16)s[j];
      }
    }
  }
  __syncthreads();

  // out = scores @ V + Q @ P_c
#pragma unroll
  for (int mt = 0; mt < 4; ++mt) {
#pragma unroll
    for (int nt = 0; nt < 2; ++nt) {
      v8f acc = vz;
      v16bf bp = load_B_rm_f32(pptr + nt * 16, K_);   // P_c: 32x32 f32 rm
      acc = wmma_bf16(aq[mt], bp, acc);
#pragma unroll
      for (int kt = 0; kt < 2; ++kt) {
        v16bf as = frag_row32(&S[mt * 16 + n][kt * 32], g);
        v16bf bv = load_B_rm_bf16(vptr + (size_t)kt * 32 * K_ + nt * 16, K_);
        acc = wmma_bf16(as, bv, acc);
      }
      __bf16* dst =
          attn + ((size_t)b * T_ + t0 + mt * 16) * HK_ + h * K_ + nt * 16;
      store_D_bf16(dst, HK_, acc);
    }
  }
}

// ---------------------------------------------------------------------------
// Kernel 5: out = attn @ W_out^T + b_out   (f32 output)
// ---------------------------------------------------------------------------
__global__ __launch_bounds__(256) void k_out(const __bf16* __restrict__ attn,
                                             const __bf16* __restrict__ Wo,
                                             const float* __restrict__ bout,
                                             float* __restrict__ out) {
  __shared__ __align__(16) __bf16 Abf[2][64][40];
  __shared__ __align__(16) __bf16 Bbf[2][128][40];
  const int m0 = blockIdx.x * 64;
  const int n0 = blockIdx.y * 128;
  const int tid = threadIdx.x;
  const int wave = tid >> 5, lane = tid & 31;
  const int m = lane & 15, g = lane >> 4;
  const int mt = wave & 3;
  const int ntBase = (wave >> 2) * 4;
  const int ar = tid >> 2, ac = (tid & 3) * 8;
  const v8f vz = {0.f, 0.f, 0.f, 0.f, 0.f, 0.f, 0.f, 0.f};
  v8f acc[4] = {vz, vz, vz, vz};
  const int NK = HK_ / 32;

#if HAVE_ASYNC
  auto issue_tile = [&](int ks) {
    int kk = ks * 32, bufi = ks & 1;
    GASYNC_B128(attn + (size_t)(m0 + ar) * HK_ + kk + ac, &Abf[bufi][ar][ac]);
#pragma unroll
    for (int i = 0; i < 2; ++i) {
      int cidx = tid + 256 * i;
      int r = cidx >> 2, c8 = (cidx & 3) * 8;
      GASYNC_B128(Wo + (size_t)(n0 + r) * HK_ + kk + c8, &Bbf[bufi][r][c8]);
    }
  };
#else
  auto copy_tile = [&](int ks) {
    int kk = ks * 32, bufi = ks & 1;
    *(v8bf*)&Abf[bufi][ar][ac] =
        *(const v8bf*)(attn + (size_t)(m0 + ar) * HK_ + kk + ac);
#pragma unroll
    for (int i = 0; i < 2; ++i) {
      int cidx = tid + 256 * i;
      int r = cidx >> 2, c8 = (cidx & 3) * 8;
      *(v8bf*)&Bbf[bufi][r][c8] =
          *(const v8bf*)(Wo + (size_t)(n0 + r) * HK_ + kk + c8);
    }
  };
#endif
  auto compute_tile = [&](int bufi) {
    v16bf a = frag_row32(&Abf[bufi][mt * 16 + m][0], g);
#pragma unroll
    for (int t4 = 0; t4 < 4; ++t4) {
      v16bf b = frag_row32(&Bbf[bufi][(ntBase + t4) * 16 + m][0], g);
      acc[t4] = wmma_bf16(a, b, acc[t4]);
    }
  };

#if HAVE_ASYNC
  issue_tile(0);
  for (int ks = 0; ks < NK; ++ks) {
    if (ks + 1 < NK) { issue_tile(ks + 1); WAIT_ASYNC(3); } else { WAIT_ASYNC(0); }
    __syncthreads();
    compute_tile(ks & 1);
    __syncthreads();
  }
#else
  for (int ks = 0; ks < NK; ++ks) {
    copy_tile(ks);
    __syncthreads();
    compute_tile(ks & 1);
    __syncthreads();
  }
#endif

#pragma unroll
  for (int t4 = 0; t4 < 4; ++t4) {
    int nt = ntBase + t4;
#pragma unroll
    for (int j = 0; j < 8; ++j) {
      int gm = m0 + mt * 16 + j + 8 * g;
      int gn = n0 + nt * 16 + m;
      out[(size_t)gm * C_ + gn] = acc[t4][j] + bout[gn];
    }
  }
}

// ---------------------------------------------------------------------------
extern "C" void kernel_launch(void* const* d_in, const int* in_sizes, int n_in,
                              void* d_out, int out_size, void* d_ws, size_t ws_size,
                              hipStream_t stream) {
  const float* x     = (const float*)d_in[0];
  const float* sbas  = (const float*)d_in[1];
  // d_in[2] = spectral_freqs: unused by the reference computation
  const float* Wqkv  = (const float*)d_in[3];
  const float* Wout  = (const float*)d_in[4];
  const float* bout  = (const float*)d_in[5];
  float* out = (float*)d_out;

  // workspace: bf16 copies + bf16 intermediates + f32 chunk states (~38 MB)
  __bf16* xb  = (__bf16*)d_ws;                    // 8,388,608
  __bf16* Wqb = xb + (size_t)MROWS_ * C_;         //   786,432
  __bf16* Wob = Wqb + (size_t)NQKV_ * C_;         //   262,144
  const size_t QKV_E = (size_t)BH_ * T_ * K_;     // 2,097,152
  __bf16* qb    = Wob + (size_t)C_ * HK_;
  __bf16* kb    = qb + QKV_E;
  __bf16* vb    = kb + QKV_E;
  __bf16* attnb = vb + QKV_E;
  float*  Sb    = (float*)(attnb + QKV_E);        // 1,048,576 f32

  k_cvt<<<(MROWS_ * C_ / 8) / 256, 256, 0, stream>>>(x, xb, MROWS_ * C_ / 8);
  k_cvt<<<(NQKV_ * C_ / 8) / 256, 256, 0, stream>>>(Wqkv, Wqb, NQKV_ * C_ / 8);
  k_cvt<<<(C_ * HK_ / 8) / 256, 256, 0, stream>>>(Wout, Wob, C_ * HK_ / 8);

  k_qkv  <<<dim3(MROWS_ / 64, NQKV_ / 128), 256, 0, stream>>>(xb, sbas, Wqb,
                                                              qb, kb, vb);
  k_state<<<(BH_ * NC_) / 8, 256, 0, stream>>>(kb, vb, Sb);
  k_prefix<<<BH_, 256, 0, stream>>>(Sb);
  k_attn <<<(BH_ * NC_) / 4, 128, 0, stream>>>(qb, kb, vb, Sb, attnb);
  k_out  <<<dim3(MROWS_ / 64, C_ / 128), 256, 0, stream>>>(attnb, Wob, bout, out);
}
